// LSTM_23802708755226
// MI455X (gfx1250) — matrix operand order repaired and checked by reading, not compiled, not run on hardware
//
#include <hip/hip_runtime.h>
#include <hip/hip_bf16.h>
#include <math.h>

// ---------------------------------------------------------------------------
// Types for CDNA5 WMMA (wave32): v_wmma_f32_16x16x32_bf16
// ---------------------------------------------------------------------------
typedef __attribute__((ext_vector_type(16))) __bf16 v16bf;
typedef __attribute__((ext_vector_type(8)))  float  v8f;
typedef __attribute__((ext_vector_type(4)))  unsigned int u32x4;

union FragBF {
  v16bf v;
  u32x4 q[2];
};

static __device__ __forceinline__ v8f vzero8() {
  v8f z;
#pragma unroll
  for (int i = 0; i < 8; ++i) z[i] = 0.0f;
  return z;
}

static __device__ __forceinline__ float sigmoidf_fast(float x) {
  return 1.0f / (1.0f + __expf(-x));
}

static __device__ __forceinline__ v8f wmma_bf16(v16bf a, v16bf b, v8f c) {
  return __builtin_amdgcn_wmma_f32_16x16x32_bf16(false, a, false, b, (short)0, c,
                                                 false, false);
}

// A fragment (16x32 bf16, M x K), contiguous K order (matches B):
//   lane L (L<16):  row L,     K = {k0..k0+15}
//   lane L (L>=16): row L-16,  K = {k0+16..k0+31}
// -> two contiguous 16B loads per lane.
static __device__ __forceinline__ v16bf load_a_frag(const __bf16* __restrict__ A,
                                                    int lda, int row, int k0, int lane) {
  int kh = (lane >> 4) * 16;
  const __bf16* r = A + (size_t)row * lda + k0 + kh;
  FragBF f;
  f.q[0] = *reinterpret_cast<const u32x4*>(r);
  f.q[1] = *reinterpret_cast<const u32x4*>(r + 8);
  return f.v;
}

// B fragment (32x16 bf16, K x N) built from row-major W[N][K] (B = W^T):
//   lane L: column n = n0 + (L&15), K range = k0 + (L>>4)*16 .. +15
// -> one contiguous 32B load per lane (two 16B chunks).
static __device__ __forceinline__ v16bf load_b_frag(const __bf16* __restrict__ W,
                                                    int ldw, int n0, int k0, int lane) {
  int n  = n0 + (lane & 15);
  int kb = k0 + ((lane >> 4) << 4);
  const __bf16* r = W + (size_t)n * ldw + kb;
  FragBF f;
  f.q[0] = reinterpret_cast<const u32x4*>(r)[0];
  f.q[1] = reinterpret_cast<const u32x4*>(r)[1];
  return f.v;
}

// ---------------------------------------------------------------------------
// Elementwise helpers
// ---------------------------------------------------------------------------
__global__ void k_convert_bf16(const float* __restrict__ src, __bf16* __restrict__ dst, int n) {
  for (int i = blockIdx.x * blockDim.x + threadIdx.x; i < n; i += gridDim.x * blockDim.x)
    dst[i] = (__bf16)src[i];
}

__global__ void k_bias_sum(const float* __restrict__ a, const float* __restrict__ b,
                           float* __restrict__ out, int n) {
  int i = blockIdx.x * blockDim.x + threadIdx.x;
  if (i < n) out[i] = a[i] + b[i];
}

// Gather embeddings for concatenated (context || targets) sequence.
// X[b*512 + t][e], t<256 -> context, else targets. B=8, T=512, E=256.
__global__ void k_embed_gather(const int* __restrict__ ctx, const int* __restrict__ tgt,
                               const float* __restrict__ emb, __bf16* __restrict__ X) {
  int idx = blockIdx.x * blockDim.x + threadIdx.x;
  if (idx >= 4096 * 256) return;
  int row = idx >> 8, col = idx & 255;
  int b = row >> 9, t = row & 511;
  int id = (t < 256) ? ctx[b * 256 + t] : tgt[b * 256 + (t - 256)];
  X[idx] = (__bf16)emb[(size_t)id * 256 + col];
}

// ---------------------------------------------------------------------------
// WMMA GEMM: C[M,N] = A[M,K] @ W[N,K]^T + bias[N]
// block = 128 threads = 4 waves; each wave computes a 32x64 tile
// (2 M-subtiles x 4 N-subtiles, B fragments reused across both M rows).
// block tile = 32 x 256. Double-buffered K loop (software pipelined).
// remap!=0: decode-row mapping for the FC GEMM
// (out row m = b*256+t -> y1 row b*512 + 256 + t).
// ---------------------------------------------------------------------------
__global__ __launch_bounds__(128) void k_gemm_wmma(
    const __bf16* __restrict__ A, int lda,
    const __bf16* __restrict__ W, int ldw,
    const float* __restrict__ bias, float* __restrict__ C,
    int N, int K, int remap)
{
  int tid = threadIdx.x;
  int lane = tid & 31;
  int w = tid >> 5;                       // 0..3
  int m0 = blockIdx.y * 32;
  int n0 = blockIdx.x * 256 + w * 64;

  // per-lane A rows for the two M-subtiles (with optional decode remap)
  int am0 = m0 + (lane & 15);
  int am1 = am0 + 16;
  int ar0 = remap ? (((am0 >> 8) << 9) + 256 + (am0 & 255)) : am0;
  int ar1 = remap ? (((am1 >> 8) << 9) + 256 + (am1 & 255)) : am1;

  v8f acc[2][4];
#pragma unroll
  for (int u = 0; u < 2; ++u)
#pragma unroll
    for (int s = 0; s < 4; ++s) acc[u][s] = vzero8();

  // ---- prologue: fragments for k0 = 0 ----
  v16bf a0 = load_a_frag(A, lda, ar0, 0, lane);
  v16bf a1 = load_a_frag(A, lda, ar1, 0, lane);
  v16bf b0 = load_b_frag(W, ldw, n0 +  0, 0, lane);
  v16bf b1 = load_b_frag(W, ldw, n0 + 16, 0, lane);
  v16bf b2 = load_b_frag(W, ldw, n0 + 32, 0, lane);
  v16bf b3 = load_b_frag(W, ldw, n0 + 48, 0, lane);

  // ---- pipelined main loop: load k0+32 while computing k0 ----
  for (int k0 = 0; k0 + 32 < K; k0 += 32) {
    int kn = k0 + 32;
    v16bf na0 = load_a_frag(A, lda, ar0, kn, lane);
    v16bf na1 = load_a_frag(A, lda, ar1, kn, lane);
    v16bf nb0 = load_b_frag(W, ldw, n0 +  0, kn, lane);
    v16bf nb1 = load_b_frag(W, ldw, n0 + 16, kn, lane);
    v16bf nb2 = load_b_frag(W, ldw, n0 + 32, kn, lane);
    v16bf nb3 = load_b_frag(W, ldw, n0 + 48, kn, lane);

    acc[0][0] = wmma_bf16(a0, b0, acc[0][0]);
    acc[1][0] = wmma_bf16(a1, b0, acc[1][0]);
    acc[0][1] = wmma_bf16(a0, b1, acc[0][1]);
    acc[1][1] = wmma_bf16(a1, b1, acc[1][1]);
    acc[0][2] = wmma_bf16(a0, b2, acc[0][2]);
    acc[1][2] = wmma_bf16(a1, b2, acc[1][2]);
    acc[0][3] = wmma_bf16(a0, b3, acc[0][3]);
    acc[1][3] = wmma_bf16(a1, b3, acc[1][3]);

    a0 = na0; a1 = na1;
    b0 = nb0; b1 = nb1; b2 = nb2; b3 = nb3;
  }
  // ---- epilogue: last K-iteration ----
  acc[0][0] = wmma_bf16(a0, b0, acc[0][0]);
  acc[1][0] = wmma_bf16(a1, b0, acc[1][0]);
  acc[0][1] = wmma_bf16(a0, b1, acc[0][1]);
  acc[1][1] = wmma_bf16(a1, b1, acc[1][1]);
  acc[0][2] = wmma_bf16(a0, b2, acc[0][2]);
  acc[1][2] = wmma_bf16(a1, b2, acc[1][2]);
  acc[0][3] = wmma_bf16(a0, b3, acc[0][3]);
  acc[1][3] = wmma_bf16(a1, b3, acc[1][3]);

  // store: VGPR j -> row m0 + u*16 + j + 8*(lane>>4), col n0 + s*16 + (lane&15)
  int mh = (lane >> 4) * 8, nn = lane & 15;
#pragma unroll
  for (int s = 0; s < 4; ++s) {
    int n = n0 + s * 16 + nn;
    float bv = bias[n];
#pragma unroll
    for (int u = 0; u < 2; ++u)
#pragma unroll
      for (int j = 0; j < 8; ++j) {
        int m = m0 + u * 16 + mh + j;
        C[(size_t)m * N + n] = acc[u][s][j] + bv;
      }
  }
}

// ---------------------------------------------------------------------------
// Persistent single-workgroup LSTM recurrence (one layer, 512 steps).
// h kept in LDS (bf16, 16 rows: 8 batch + 8 zero-pad), c in registers.
// Per step: gates = h @ Whh^T (WMMA, 32 waves x 64 cols, double-buffered K),
// + xg[:,t,:], activations -> LDS, then fused c/h update.
//   xg : [4096, 2048] f32, row = b*512 + t
//   Whh: [2048, 512] bf16 (L2-resident, 2 MB)
//   y  : [4096, 512] bf16 output (row = b*512 + t)
// ---------------------------------------------------------------------------
#define H_LD 528  // row stride (bf16 elems), multiple of 8 -> 16B-aligned rows

__global__ __launch_bounds__(1024) void k_lstm_layer(
    const float* __restrict__ xg,
    const __bf16* __restrict__ Whh,
    __bf16* __restrict__ y)
{
  __shared__ alignas(16) __bf16 h_lds[16 * H_LD];     // ~16.5 KB
  __shared__ alignas(16) __bf16 g_lds[8 * 2048];      // 32 KB activated gates

  int tid = threadIdx.x;
  int lane = tid & 31;
  int w = tid >> 5;       // 0..31
  int n0 = w * 64;        // this wave's gate-column base (covers all 4H=2048)

  for (int i = tid; i < 16 * H_LD; i += 1024) h_lds[i] = (__bf16)0.0f;
  float c_reg[4] = {0.0f, 0.0f, 0.0f, 0.0f};
  __syncthreads();

  int mh = (lane >> 4) * 8, nn = lane & 15;

  for (int t = 0; t < 512; ++t) {
    // ---- gates tile: acc = h @ Whh^T, cols n0..n0+63 (pipelined K) ----
    v8f acc[4];
#pragma unroll
    for (int s = 0; s < 4; ++s) acc[s] = vzero8();

    v16bf af = load_a_frag(h_lds, H_LD, lane & 15, 0, lane);
    v16bf b0 = load_b_frag(Whh, 512, n0 +  0, 0, lane);
    v16bf b1 = load_b_frag(Whh, 512, n0 + 16, 0, lane);
    v16bf b2 = load_b_frag(Whh, 512, n0 + 32, 0, lane);
    v16bf b3 = load_b_frag(Whh, 512, n0 + 48, 0, lane);

    for (int k0 = 0; k0 + 32 < 512; k0 += 32) {
      int kn = k0 + 32;
      v16bf naf = load_a_frag(h_lds, H_LD, lane & 15, kn, lane);
      v16bf nb0 = load_b_frag(Whh, 512, n0 +  0, kn, lane);
      v16bf nb1 = load_b_frag(Whh, 512, n0 + 16, kn, lane);
      v16bf nb2 = load_b_frag(Whh, 512, n0 + 32, kn, lane);
      v16bf nb3 = load_b_frag(Whh, 512, n0 + 48, kn, lane);

      acc[0] = wmma_bf16(af, b0, acc[0]);
      acc[1] = wmma_bf16(af, b1, acc[1]);
      acc[2] = wmma_bf16(af, b2, acc[2]);
      acc[3] = wmma_bf16(af, b3, acc[3]);

      af = naf; b0 = nb0; b1 = nb1; b2 = nb2; b3 = nb3;
    }
    acc[0] = wmma_bf16(af, b0, acc[0]);
    acc[1] = wmma_bf16(af, b1, acc[1]);
    acc[2] = wmma_bf16(af, b2, acc[2]);
    acc[3] = wmma_bf16(af, b3, acc[3]);

    // ---- add input projection, activate, stage to LDS ----
    if (mh == 0) {  // only rows 0..7 are real batches
#pragma unroll
      for (int s = 0; s < 4; ++s) {
        int n = n0 + s * 16 + nn;
        int gate = n >> 9;  // 0:i 1:f 2:g 3:o
#pragma unroll
        for (int j = 0; j < 8; ++j) {
          int b = j;
          float v = acc[s][j] + xg[((size_t)(b * 512 + t)) * 2048 + n];
          v = (gate == 2) ? tanhf(v) : sigmoidf_fast(v);
          g_lds[b * 2048 + n] = (__bf16)v;
        }
      }
    }
    __syncthreads();

    // ---- fused cell/hidden update: 4096 (b,j) pairs over 1024 threads ----
#pragma unroll
    for (int r = 0; r < 4; ++r) {
      int flat = tid + 1024 * r;          // 0..4095
      int b = flat >> 9, j = flat & 511;
      float iv = (float)g_lds[b * 2048 + j];
      float fv = (float)g_lds[b * 2048 + 512 + j];
      float gv = (float)g_lds[b * 2048 + 1024 + j];
      float ov = (float)g_lds[b * 2048 + 1536 + j];
      c_reg[r] = fv * c_reg[r] + iv * gv;
      float hv = ov * tanhf(c_reg[r]);
      __bf16 hb = (__bf16)hv;
      h_lds[b * H_LD + j] = hb;
      y[((size_t)(b * 512 + t)) * 512 + j] = hb;
    }
    __syncthreads();
  }
}

// ---------------------------------------------------------------------------
// Host-side orchestration
// ---------------------------------------------------------------------------
extern "C" void kernel_launch(void* const* d_in, const int* in_sizes, int n_in,
                              void* d_out, int out_size, void* d_ws, size_t ws_size,
                              hipStream_t stream) {
  (void)in_sizes; (void)n_in; (void)out_size; (void)ws_size;

  const int*   context = (const int*)  d_in[0];   // [8,256]
  const int*   targets = (const int*)  d_in[1];   // [8,256]
  const float* emb     = (const float*)d_in[2];   // [32000,256]
  const float* W_ih0   = (const float*)d_in[3];   // [2048,256]
  const float* W_hh0   = (const float*)d_in[4];   // [2048,512]
  const float* b_ih0   = (const float*)d_in[5];
  const float* b_hh0   = (const float*)d_in[6];
  const float* W_ih1   = (const float*)d_in[7];   // [2048,512]
  const float* W_hh1   = (const float*)d_in[8];   // [2048,512]
  const float* b_ih1   = (const float*)d_in[9];
  const float* b_hh1   = (const float*)d_in[10];
  const float* fc_W    = (const float*)d_in[11];  // [32000,512]
  const float* fc_b    = (const float*)d_in[12];  // [32000]
  float* out = (float*)d_out;                     // [8,256,32000]

  // ---- workspace carve-up (256B aligned) ----
  char* base = (char*)d_ws;
  size_t off = 0;
  auto carve = [&](size_t bytes) -> char* {
    char* p = base + off;
    off += (bytes + 255) & ~(size_t)255;
    return p;
  };
  __bf16* wih0_bf = (__bf16*)carve((size_t)2048 * 256 * 2);
  __bf16* whh0_bf = (__bf16*)carve((size_t)2048 * 512 * 2);
  __bf16* wih1_bf = (__bf16*)carve((size_t)2048 * 512 * 2);
  __bf16* whh1_bf = (__bf16*)carve((size_t)2048 * 512 * 2);
  __bf16* fcw_bf  = (__bf16*)carve((size_t)32000 * 512 * 2);
  __bf16* X_bf    = (__bf16*)carve((size_t)4096 * 256 * 2);
  __bf16* y0_bf   = (__bf16*)carve((size_t)4096 * 512 * 2);
  __bf16* y1_bf   = (__bf16*)carve((size_t)4096 * 512 * 2);
  float*  xg      = (float*) carve((size_t)4096 * 2048 * 4);
  float*  b0      = (float*) carve((size_t)2048 * 4);
  float*  b1      = (float*) carve((size_t)2048 * 4);

  // ---- 1) weight conversions to bf16 ----
  auto conv = [&](const float* s, __bf16* d, int n) {
    int blocks = (n + 255) / 256; if (blocks > 4096) blocks = 4096;
    k_convert_bf16<<<blocks, 256, 0, stream>>>(s, d, n);
  };
  conv(W_ih0, wih0_bf, 2048 * 256);
  conv(W_hh0, whh0_bf, 2048 * 512);
  conv(W_ih1, wih1_bf, 2048 * 512);
  conv(W_hh1, whh1_bf, 2048 * 512);
  conv(fc_W,  fcw_bf,  32000 * 512);

  k_bias_sum<<<8, 256, 0, stream>>>(b_ih0, b_hh0, b0, 2048);
  k_bias_sum<<<8, 256, 0, stream>>>(b_ih1, b_hh1, b1, 2048);

  // ---- 2) embedding gather (ctx || tgt), f32 -> bf16 ----
  k_embed_gather<<<4096, 256, 0, stream>>>(context, targets, emb, X_bf);

  // ---- 3) xg0 = X @ W_ih0^T + b0 : [4096,256]x[2048,256]^T ----
  {
    dim3 grid(2048 / 256, 4096 / 32);
    k_gemm_wmma<<<grid, 128, 0, stream>>>(X_bf, 256, wih0_bf, 256, b0, xg,
                                          2048, 256, 0);
  }

  // ---- 4) layer-0 recurrence (512 steps, persistent workgroup) ----
  k_lstm_layer<<<1, 1024, 0, stream>>>(xg, whh0_bf, y0_bf);

  // ---- 5) xg1 = y0 @ W_ih1^T + b1 : [4096,512]x[2048,512]^T ----
  {
    dim3 grid(2048 / 256, 4096 / 32);
    k_gemm_wmma<<<grid, 128, 0, stream>>>(y0_bf, 512, wih1_bf, 512, b1, xg,
                                          2048, 512, 0);
  }

  // ---- 6) layer-1 recurrence ----
  k_lstm_layer<<<1, 1024, 0, stream>>>(xg, whh1_bf, y1_bf);

  // ---- 7) logits = y1[decode rows] @ fc_W^T + fc_b : [2048,512]x[32000,512]^T ----
  {
    dim3 grid(32000 / 256, 2048 / 32);
    k_gemm_wmma<<<grid, 128, 0, stream>>>(y1_bf, 512, fcw_bf, 512, fc_b, out,
                                          32000, 512, 1);
  }
}
